// EnsembleActor_60902636257569
// MI455X (gfx1250) — matrix operand order repaired and checked by reading, not compiled, not run on hardware
//
#include <hip/hip_runtime.h>
#include <hip/hip_bf16.h>
#include <math.h>
#include <stdint.h>

// Problem constants (from reference)
#define E_   8
#define B_   4096
#define OBS_ 256
#define H_   1024
#define A_   64

// LDS tile geometry: K-step 32 elements, rows padded 32 -> 40 elements (80 B)
// so 16-lane fragment reads map to distinct bank groups (i*20 mod 64 is a
// permutation of stride-4-aligned groups) -> conflict-free ds_load_b128.
#define KSTEP 32
#define PADK  40
#define MT    256   // block tile M
#define NT    128   // block tile N

typedef __bf16 bf16;
typedef __attribute__((ext_vector_type(16))) __bf16 v16bf;
typedef __attribute__((ext_vector_type(8)))  __bf16 v8bf;
typedef __attribute__((ext_vector_type(8)))  float  v8f;

// ---------------------------------------------------------------------------
// CDNA5 async global->LDS copy (ASYNCcnt-tracked, no VGPR round trip).
// VFLAT async encoding: vdst = per-lane LDS byte address, vaddr = 64-bit
// global address, saddr = off.
// ---------------------------------------------------------------------------
__device__ __forceinline__ void async_copy16(uint32_t lds_addr, const void* gaddr) {
  asm volatile("global_load_async_to_lds_b128 %0, %1, off"
               :: "v"(lds_addr), "v"((unsigned long long)(uintptr_t)gaddr)
               : "memory");
}

#if defined(__has_builtin)
#  if __has_builtin(__builtin_amdgcn_s_wait_asynccnt)
#    define WAIT_ASYNCCNT(n) __builtin_amdgcn_s_wait_asynccnt(n)
#  endif
#endif
#ifndef WAIT_ASYNCCNT
#  define WAIT_ASYNCCNT(n) asm volatile("s_wait_asynccnt " #n ::: "memory")
#endif

// ---------------------------------------------------------------------------
// WMMA fragment loaders from LDS (wave32, 16x16x32 bf16).
// A (16x32, per ISA 7.12.2): lane l<16 -> row l, K = {kb..kb+7, kb+16..kb+23},
// kb = 0; lanes 16-31 same rows, kb = 8.
// B (32x16, weights pre-transposed Wt[N,K]): lane l<16 -> col l, K = {0..7, 16..23};
// lanes 16-31 -> K = {8..15, 24..31}.
// ---------------------------------------------------------------------------
__device__ __forceinline__ v16bf lds_frag_a(const bf16* buf, int row) {
  const int lane = threadIdx.x & 31;
  const bf16* p = buf + (row + (lane & 15)) * PADK + ((lane >> 4) << 3);
  v8bf lo = *reinterpret_cast<const v8bf*>(p);
  v8bf hi = *reinterpret_cast<const v8bf*>(p + 16);
  v16bf f;
#pragma unroll
  for (int i = 0; i < 8; ++i) { f[i] = lo[i]; f[i + 8] = hi[i]; }
  return f;
}

__device__ __forceinline__ v16bf lds_frag_b(const bf16* buf, int col) {
  const int lane = threadIdx.x & 31;
  const bf16* p = buf + (col + (lane & 15)) * PADK + ((lane >> 4) << 3);
  v8bf lo = *reinterpret_cast<const v8bf*>(p);
  v8bf hi = *reinterpret_cast<const v8bf*>(p + 16);
  v16bf f;
#pragma unroll
  for (int i = 0; i < 8; ++i) { f[i] = lo[i]; f[i + 8] = hi[i]; }
  return f;
}

// Direct-from-global fragment loaders (used by the small head GEMM).
__device__ __forceinline__ v16bf glb_frag_a(const bf16* __restrict__ base,
                                            int row, int ld, int k) {
  const int lane = threadIdx.x & 31;
  const bf16* p = base + (size_t)(row + (lane & 15)) * ld + k + ((lane >> 4) << 3);
  v8bf lo = *reinterpret_cast<const v8bf*>(p);
  v8bf hi = *reinterpret_cast<const v8bf*>(p + 16);
  v16bf f;
#pragma unroll
  for (int i = 0; i < 8; ++i) { f[i] = lo[i]; f[i + 8] = hi[i]; }
  return f;
}
__device__ __forceinline__ v16bf glb_frag_b(const bf16* __restrict__ base,
                                            int col, int ld, int k) {
  const int lane = threadIdx.x & 31;
  const bf16* p = base + (size_t)(col + (lane & 15)) * ld + k + ((lane >> 4) << 3);
  v8bf lo = *reinterpret_cast<const v8bf*>(p);
  v8bf hi = *reinterpret_cast<const v8bf*>(p + 16);
  v16bf f;
#pragma unroll
  for (int i = 0; i < 8; ++i) { f[i] = lo[i]; f[i + 8] = hi[i]; }
  return f;
}

// ---------------------------------------------------------------------------
// Converters
// ---------------------------------------------------------------------------
__global__ void k_f32_to_bf16(const float* __restrict__ src,
                              bf16* __restrict__ dst, int n) {
  int i = blockIdx.x * blockDim.x + threadIdx.x;
  if (i < n) dst[i] = (bf16)src[i];
}

// fp32 W[e][K][N] -> bf16 Wt[e][N][K]  (convert + transpose, coalesced read)
__global__ void k_transpose_cvt(const float* __restrict__ W,
                                bf16* __restrict__ Wt,
                                int K, int N, int total) {
  int i = blockIdx.x * blockDim.x + threadIdx.x;
  if (i >= total) return;
  int kn = K * N;
  int e  = i / kn;
  int r  = i - e * kn;
  int k  = r / N;
  int nn = r - k * N;
  Wt[((size_t)e * N + nn) * K + k] = (bf16)W[i];
}

// ---------------------------------------------------------------------------
// C = relu(A x W + b) per ensemble, async-LDS double-buffered WMMA GEMM.
// Block = 256 thr = 8 wave32, block tile 256x128, wave tile 64x64 (4x4 frags).
// Each K-step: stage A(256x32) + B(128x32) via 6 async b128 copies/thread,
// then 16 v_wmma per wave from conflict-free LDS fragments.
// ---------------------------------------------------------------------------
__global__ __launch_bounds__(256)
void k_gemm_relu(const bf16* __restrict__ Aact, const bf16* __restrict__ Wt,
                 const float* __restrict__ bias, bf16* __restrict__ Cout,
                 int K, int N) {
  __shared__ __align__(16) bf16 sA[2][MT * PADK];   // 2 x 20 KB
  __shared__ __align__(16) bf16 sB[2][NT * PADK];   // 2 x 10 KB

  const int e    = blockIdx.z;
  const int m0   = blockIdx.x * MT;
  const int n0   = blockIdx.y * NT;
  const int w    = threadIdx.x >> 5;
  const int wm   = (w & 3) * 64;      // 4 waves tile M
  const int wn   = (w >> 2) * 64;     // 2 waves tile N
  const int lane = threadIdx.x & 31;
  const int half = lane >> 4;
  const int t    = threadIdx.x;

  const bf16*  Ae = Aact + (size_t)e * B_ * K;
  const bf16*  We = Wt   + (size_t)e * N  * K;
  const float* be = bias + (size_t)e * N;
  bf16*        Ce = Cout + (size_t)e * B_ * N;

  // Per-thread staging slots (6 async ops / thread / K-step)
  const bf16*   gA  = Ae + (size_t)(m0 + t) * K;           // A row t
  const uint32_t lA0 = (uint32_t)(uintptr_t)(&sA[0][t * PADK]);
  const uint32_t lA1 = (uint32_t)(uintptr_t)(&sA[1][t * PADK]);
  const int     brow = t >> 1;
  const int     bh   = (t & 1) * 16;                       // element offset
  const bf16*   gB  = We + (size_t)(n0 + brow) * K + bh;
  const uint32_t lB0 = (uint32_t)(uintptr_t)(&sB[0][brow * PADK + bh]);
  const uint32_t lB1 = (uint32_t)(uintptr_t)(&sB[1][brow * PADK + bh]);

  v8f acc[4][4];
#pragma unroll
  for (int mi = 0; mi < 4; ++mi)
#pragma unroll
    for (int ni = 0; ni < 4; ++ni)
#pragma unroll
      for (int j = 0; j < 8; ++j) acc[mi][ni][j] = 0.0f;

  // Prologue: stage K-step 0 into buffer 0
  {
#pragma unroll
    for (int j = 0; j < 4; ++j) async_copy16(lA0 + j * 16, gA + j * 8);
    async_copy16(lB0,      gB);
    async_copy16(lB0 + 16, gB + 8);
  }

  const int nIter = K / KSTEP;
  for (int i = 0; i < nIter; ++i) {
    const int cur = i & 1;
    if (i + 1 < nIter) {
      // Prefetch next K-step into the other buffer (last read in iter i-1,
      // protected by the trailing barrier of iter i-1).
      const int kk = (i + 1) * KSTEP;
      const uint32_t la = cur ? lA0 : lA1;
      const uint32_t lb = cur ? lB0 : lB1;
#pragma unroll
      for (int j = 0; j < 4; ++j) async_copy16(la + j * 16, gA + kk + j * 8);
      async_copy16(lb,      gB + kk);
      async_copy16(lb + 16, gB + kk + 8);
      WAIT_ASYNCCNT(6);          // older (current-buffer) copies complete
    } else {
      WAIT_ASYNCCNT(0);
    }
    __syncthreads();             // everyone's current-buffer data is in LDS

    v16bf a[4], b[4];
#pragma unroll
    for (int mi = 0; mi < 4; ++mi) a[mi] = lds_frag_a(sA[cur], wm + mi * 16);
#pragma unroll
    for (int ni = 0; ni < 4; ++ni) b[ni] = lds_frag_b(sB[cur], wn + ni * 16);
#pragma unroll
    for (int mi = 0; mi < 4; ++mi)
#pragma unroll
      for (int ni = 0; ni < 4; ++ni)
        acc[mi][ni] = __builtin_amdgcn_wmma_f32_16x16x32_bf16(
            false, a[mi], false, b[ni], (short)0, acc[mi][ni], false, false);

    __syncthreads();             // all reads done before buffer reuse
  }

  // Epilogue: +bias, ReLU, bf16 store. C layout: lane<16 -> M=j, lane>=16 -> M=j+8.
#pragma unroll
  for (int mi = 0; mi < 4; ++mi) {
#pragma unroll
    for (int ni = 0; ni < 4; ++ni) {
      const int col = n0 + wn + ni * 16 + (lane & 15);
      const float bv = be[col];
#pragma unroll
      for (int j = 0; j < 8; ++j) {
        const int row = m0 + wm + mi * 16 + j + half * 8;
        float v = acc[mi][ni][j] + bv;
        v = v > 0.0f ? v : 0.0f;
        Ce[(size_t)row * N + col] = (bf16)v;
      }
    }
  }
}

// ---------------------------------------------------------------------------
// Final layer + head: mu = h2 x W3 + b3 ; Gs = mean|mu| clamp>=1 ; mu/=Gs ;
// pi = mu + 0.1*noise ; out = tanh. N = A_ = 64 per wave, so the row
// reduction stays in-register + cross-lane shuffles. W3t is 1 MB: L2-resident,
// direct global fragment loads are fine here.
// ---------------------------------------------------------------------------
__global__ __launch_bounds__(256)
void k_gemm3_head(const bf16* __restrict__ h2, const bf16* __restrict__ W3t,
                  const float* __restrict__ b3, const float* __restrict__ noise,
                  float* __restrict__ out_mu, float* __restrict__ out_pi) {
  const int e    = blockIdx.z;
  const int m0   = blockIdx.x * 128 + (threadIdx.x >> 5) * 16;
  const int lane = threadIdx.x & 31;
  const int half = lane >> 4;

  const bf16* Ae = h2  + (size_t)e * B_ * H_;
  const bf16* We = W3t + (size_t)e * A_ * H_;

  v8f acc[4];
#pragma unroll
  for (int nf = 0; nf < 4; ++nf)
#pragma unroll
    for (int j = 0; j < 8; ++j) acc[nf][j] = 0.0f;

  for (int kk = 0; kk < H_; kk += KSTEP) {
    v16bf a   = glb_frag_a(Ae, m0, H_, kk);
    v16bf b0  = glb_frag_b(We,  0, H_, kk);
    v16bf b1  = glb_frag_b(We, 16, H_, kk);
    v16bf b2  = glb_frag_b(We, 32, H_, kk);
    v16bf b3f = glb_frag_b(We, 48, H_, kk);
    acc[0] = __builtin_amdgcn_wmma_f32_16x16x32_bf16(false, a, false, b0, (short)0, acc[0], false, false);
    acc[1] = __builtin_amdgcn_wmma_f32_16x16x32_bf16(false, a, false, b1, (short)0, acc[1], false, false);
    acc[2] = __builtin_amdgcn_wmma_f32_16x16x32_bf16(false, a, false, b2, (short)0, acc[2], false, false);
    acc[3] = __builtin_amdgcn_wmma_f32_16x16x32_bf16(false, a, false, b3f,(short)0, acc[3], false, false);
  }

  float bv[4];
#pragma unroll
  for (int nf = 0; nf < 4; ++nf)
    bv[nf] = b3[(size_t)e * A_ + nf * 16 + (lane & 15)];

#pragma unroll
  for (int j = 0; j < 8; ++j) {
    float s = 0.0f;
#pragma unroll
    for (int nf = 0; nf < 4; ++nf) {
      float v = acc[nf][j] + bv[nf];
      acc[nf][j] = v;
      s += fabsf(v);
    }
    // reduce across the 16 lanes of this half-wave (masks < 16 stay in-half)
    s += __shfl_xor(s, 1, 32);
    s += __shfl_xor(s, 2, 32);
    s += __shfl_xor(s, 4, 32);
    s += __shfl_xor(s, 8, 32);

    const float g   = fmaxf(s * (1.0f / (float)A_), 1.0f);
    const float inv = 1.0f / g;
    const int   row = m0 + j + half * 8;

#pragma unroll
    for (int nf = 0; nf < 4; ++nf) {
      const int col = nf * 16 + (lane & 15);
      const size_t idx = ((size_t)e * B_ + row) * A_ + col;
      const float mu = acc[nf][j] * inv;
      const float pi = mu + 0.1f * noise[idx];
      out_mu[idx] = tanhf(mu);   // ACT_LIMIT = 1.0
      out_pi[idx] = tanhf(pi);
    }
  }
}

// ---------------------------------------------------------------------------
extern "C" void kernel_launch(void* const* d_in, const int* in_sizes, int n_in,
                              void* d_out, int out_size, void* d_ws, size_t ws_size,
                              hipStream_t stream) {
  (void)in_sizes; (void)n_in; (void)out_size; (void)ws_size;

  const float* x     = (const float*)d_in[0];
  const float* noise = (const float*)d_in[1];
  const float* W1    = (const float*)d_in[2];
  const float* b1    = (const float*)d_in[3];
  const float* W2    = (const float*)d_in[4];
  const float* b2    = (const float*)d_in[5];
  const float* W3    = (const float*)d_in[6];
  const float* b3    = (const float*)d_in[7];

  float* out_mu = (float*)d_out;
  float* out_pi = out_mu + (size_t)E_ * B_ * A_;

  // Workspace layout (bf16 elements): ~165 MB total
  bf16* ws   = (bf16*)d_ws;
  size_t o   = 0;
  bf16* x_bf = ws + o; o += (size_t)E_ * B_ * OBS_;   // 8.4M el
  bf16* W1t  = ws + o; o += (size_t)E_ * H_ * OBS_;   // 2.1M
  bf16* W2t  = ws + o; o += (size_t)E_ * H_ * H_;     // 8.4M
  bf16* W3t  = ws + o; o += (size_t)E_ * A_ * H_;     // 0.5M
  bf16* h1   = ws + o; o += (size_t)E_ * B_ * H_;     // 33.6M
  bf16* h2   = ws + o;                                // 33.6M

  int n;
  n = E_ * B_ * OBS_;
  k_f32_to_bf16<<<(n + 255) / 256, 256, 0, stream>>>(x, x_bf, n);
  n = E_ * OBS_ * H_;
  k_transpose_cvt<<<(n + 255) / 256, 256, 0, stream>>>(W1, W1t, OBS_, H_, n);
  n = E_ * H_ * H_;
  k_transpose_cvt<<<(n + 255) / 256, 256, 0, stream>>>(W2, W2t, H_, H_, n);
  n = E_ * H_ * A_;
  k_transpose_cvt<<<(n + 255) / 256, 256, 0, stream>>>(W3, W3t, H_, A_, n);

  dim3 blk(256);
  dim3 g12(B_ / MT, H_ / NT, E_);
  k_gemm_relu<<<g12, blk, 0, stream>>>(x_bf, W1t, b1, h1, OBS_, H_);
  k_gemm_relu<<<g12, blk, 0, stream>>>(h1,   W2t, b2, h2, H_,   H_);

  dim3 g3(B_ / 128, 1, E_);
  k_gemm3_head<<<g3, blk, 0, stream>>>(h2, W3t, b3, noise, out_mu, out_pi);
}
